// ShortTermLSTM_40389872452132
// MI455X (gfx1250) — compile-verified, matrix operand order checked by reading, not run on hardware
//
#include <hip/hip_runtime.h>
#include <hip/hip_bf16.h>
#include <stdint.h>

typedef __attribute__((ext_vector_type(16))) __bf16 v16bf;
typedef __attribute__((ext_vector_type(8)))  float  v8f;
typedef __attribute__((ext_vector_type(2)))  float  v2f;

#define H        100
#define T_STEPS  128
#define BM       16        // sequences per workgroup
#define NT       25        // 400 gates / 16
#define NWAVES   5         // 5 waves x 5 tiles = 25 (perfectly uniform)
#define TPW      5         // tiles per wave
#define NTHREADS 160
#define HS_STR   132       // LDS row stride (shorts) for h_bf16 (bank-conflict pad)
#define G_STR    404       // LDS row stride (floats) for gates

__device__ __forceinline__ __bf16 f2bf(float f) {
    unsigned u = __builtin_bit_cast(unsigned, f);
    unsigned r = u + 0x7FFFu + ((u >> 16) & 1u);   // round-to-nearest-even
    unsigned short s = (unsigned short)(r >> 16);
    return __builtin_bit_cast(__bf16, s);
}

#if __has_builtin(__builtin_amdgcn_tanhf)
__device__ __forceinline__ float tanh_f(float x) { return __builtin_amdgcn_tanhf(x); }
#else
__device__ __forceinline__ float tanh_f(float x) {
    float ax = fabsf(x);
    float e  = __expf(-2.0f * ax);
    float t  = (1.0f - e) / (1.0f + e);
    return (x < 0.0f) ? -t : t;
}
#endif
__device__ __forceinline__ float sig_f(float x) { return 0.5f * tanh_f(0.5f * x) + 0.5f; }

__global__ __launch_bounds__(NTHREADS) void lstm_wmma_kernel(
    const float* __restrict__ x,     // [B, T, 4]
    const float* __restrict__ Wih,   // [400, 4]
    const float* __restrict__ Whh,   // [400, 100]
    const float* __restrict__ bih,   // [400]
    const float* __restrict__ bhh,   // [400]
    float* __restrict__ out)         // [B, 100] flat (== [n_cells, 300])
{
    __shared__ unsigned short hsh[BM][HS_STR];  // h in bf16, cols >= H stay zero
    __shared__ float csh[BM * H];               // cell state
    __shared__ float hsf[BM * H];               // h in f32 (for final output)
    __shared__ float G[BM][G_STR];              // gate pre-activations

    const int tid  = threadIdx.x;
    const int wave = tid >> 5;
    const int nl   = tid & 15;          // N index within tile / M row for A
    const int half = (tid >> 4) & 1;    // lane group (K-half selector)
    const int b0   = blockIdx.x * BM;

    // ---- preload per-wave constant fragments (held in VGPRs for all 128 steps)
    // tile = wave + NWAVES*i, i in [0,5): every wave gets exactly 5 tiles.
    v16bf Bh[TPW][4];   // W_hh^T bf16 B-fragments: [tile][kstep]
    v2f   Bx[TPW];      // W_ih^T f32 B-fragments (K=4)
    float bias[TPW];
#pragma unroll
    for (int i = 0; i < TPW; ++i) {
        int tile = wave + NWAVES * i;
        int ng   = tile * 16 + nl;                      // gate column (N)
        bias[i]  = bih[ng] + bhh[ng];
        Bx[i]    = *(const v2f*)(Wih + ng * 4 + 2 * half);
#pragma unroll
        for (int ks = 0; ks < 4; ++ks) {
            int k0 = ks * 32;
            v16bf f;
#pragma unroll
            for (int v = 0; v < 8; ++v) {
                int kb = (v < 4 ? 2 * v : 8 + 2 * v) + 8 * half;
                int k  = k0 + kb;
                float w0 = (k     < H) ? Whh[(size_t)ng * H + k]     : 0.0f;
                float w1 = (k + 1 < H) ? Whh[(size_t)ng * H + k + 1] : 0.0f;
                f[2 * v]     = f2bf(w0);
                f[2 * v + 1] = f2bf(w1);
            }
            Bh[i][ks] = f;
        }
    }

    // ---- zero-init recurrent state (incl. bf16 K padding)
    for (int idx = tid; idx < BM * HS_STR; idx += NTHREADS)
        ((unsigned short*)hsh)[idx] = 0;
    for (int idx = tid; idx < BM * H; idx += NTHREADS)
        csh[idx] = 0.0f;
    __syncthreads();

    // elementwise mapping: 160 threads x 10 cells = 1600, no per-iter division
    const int m_ew = tid / 10;          // row 0..15
    const int j_ew = tid - m_ew * 10;   // col base 0..9

    const float* xrow = x + ((size_t)(b0 + nl) * T_STEPS) * 4 + 2 * half;
    v2f ax = *(const v2f*)(xrow);       // t = 0 input fragment

    // ---- time loop
    for (int t = 0; t < T_STEPS; ++t) {
        // prefetch next timestep's input fragment (overlaps elementwise phase)
        int tn = (t + 1 < T_STEPS) ? t + 1 : t;
        v2f ax_next = *(const v2f*)(xrow + (size_t)tn * 4);

        // hidden A fragments from LDS (bf16), shared across this wave's 5 tiles
        v16bf Ah[4];
#pragma unroll
        for (int ks = 0; ks < 4; ++ks) {
            int k0 = ks * 32;
            v16bf a;
#pragma unroll
            for (int v = 0; v < 8; ++v) {
                int kb = (v < 4 ? 2 * v : 8 + 2 * v) + 8 * half;
                unsigned p = *(const unsigned*)&hsh[nl][k0 + kb];
                a[2 * v]     = __builtin_bit_cast(__bf16, (unsigned short)(p & 0xFFFFu));
                a[2 * v + 1] = __builtin_bit_cast(__bf16, (unsigned short)(p >> 16));
            }
            Ah[ks] = a;
        }

        // 5 independent WMMA accumulation chains (no branches -> interleavable)
#pragma unroll
        for (int i = 0; i < TPW; ++i) {
            int tile = wave + NWAVES * i;
            v8f c;
#pragma unroll
            for (int r = 0; r < 8; ++r) c[r] = bias[i];      // bias depends only on N
            // gates += x_t @ W_ih^T   (K = 4, f32)
            c = __builtin_amdgcn_wmma_f32_16x16x4_f32(
                    false, ax, false, Bx[i], (short)0, c, false, false);
            // gates += h @ W_hh^T     (K = 128 padded, bf16)
#pragma unroll
            for (int ks = 0; ks < 4; ++ks)
                c = __builtin_amdgcn_wmma_f32_16x16x32_bf16(
                        false, Ah[ks], false, Bh[i][ks], (short)0, c, false, false);
            int tnn = tile * 16 + nl;
#pragma unroll
            for (int r = 0; r < 8; ++r)
                G[r + 8 * half][tnn] = c[r];
        }
        __syncthreads();

        // elementwise LSTM cell update (torch gate order i, f, g, o)
#pragma unroll
        for (int k = 0; k < 10; ++k) {
            int j   = j_ew + 10 * k;
            int idx = m_ew * H + j;
            float gi = G[m_ew][j];
            float gf = G[m_ew][H + j];
            float gg = G[m_ew][2 * H + j];
            float go = G[m_ew][3 * H + j];
            float cc = csh[idx];
            cc = sig_f(gf) * cc + sig_f(gi) * tanh_f(gg);
            float hh = sig_f(go) * tanh_f(cc);
            csh[idx] = cc;
            hsf[idx] = hh;
            hsh[m_ew][j] = __builtin_bit_cast(unsigned short, f2bf(hh));
        }
        __syncthreads();

        ax = ax_next;
    }

    // ---- final hidden state: out flat layout [B,100] == [n_cells, 300]
    for (int idx = tid; idx < BM * H; idx += NTHREADS)
        out[(size_t)b0 * H + idx] = hsf[idx];
}

extern "C" void kernel_launch(void* const* d_in, const int* in_sizes, int n_in,
                              void* d_out, int out_size, void* d_ws, size_t ws_size,
                              hipStream_t stream) {
    const float* x    = (const float*)d_in[0];  // short_ttf [4096,3,128,4]
    const float* Wih  = (const float*)d_in[1];  // [400,4]
    const float* Whh  = (const float*)d_in[2];  // [400,100]
    const float* bih  = (const float*)d_in[3];  // [400]
    const float* bhh  = (const float*)d_in[4];  // [400]
    float* out = (float*)d_out;

    int B_total = in_sizes[0] / (T_STEPS * 4);  // 12288 sequences
    int grid = B_total / BM;                    // 768 workgroups
    lstm_wmma_kernel<<<grid, NTHREADS, 0, stream>>>(x, Wih, Whh, bih, bhh, out);
}
